// OvOAttention_68453188763872
// MI455X (gfx1250) — matrix operand order repaired and checked by reading, not compiled, not run on hardware
//
#include <hip/hip_runtime.h>

typedef __bf16 bf16;
typedef __attribute__((ext_vector_type(16))) __bf16        v16bf;
typedef __attribute__((ext_vector_type(8)))  float         v8f;
typedef __attribute__((ext_vector_type(4)))  unsigned int  u32x4;
typedef __attribute__((ext_vector_type(4)))  __bf16        bf16x4;

#define B_DIM 1024
#define H_DIM 64
#define E_DIM 512

// padded LDS row strides (elements) -> bank-conflict-free ds_load_b128 frags
#define RS    520   // m, q, tmp rows of 512 bf16 (byte stride 1040, 16B-aligned)
#define QT_RS 72    // qT rows of 64 bf16 (byte stride 144, 16B-aligned)
#define AT_RS 72    // attn rows of 64 bf16
#define SC_RS 65    // score rows of 64 f32 (conflict-free softmax)

// LDS byte offsets
#define LDS_M     0
#define LDS_Q     66560
#define LDS_TMP   133120
#define LDS_QT    199680
#define LDS_SC    273408
#define LDS_ATT   290048
#define LDS_TOTAL 299264

__device__ __forceinline__ v16bf load_frag(const bf16* lo, const bf16* hi) {
  union { v16bf v; u32x4 u[2]; } f;
  f.u[0] = *(const u32x4*)lo;   // K chunk 0 -> VGPR 0..3
  f.u[1] = *(const u32x4*)hi;   // K chunk 1 -> VGPR 4..7
  return f.v;
}

// --- prekernel: Wt[n*512 + f] = bf16(W[f*512 + n])  (column-major-ized W) ---
__global__ __launch_bounds__(256) void wt_conv_kernel(const float* __restrict__ W,
                                                      bf16* __restrict__ Wt) {
  int idx = blockIdx.x * 256 + threadIdx.x;   // 0 .. 512*512-1
  int n = idx >> 9;
  int f = idx & (E_DIM - 1);
  Wt[idx] = (bf16)W[f * E_DIM + n];
}

// --- main kernel: one block per batch b, 256 threads = 8 wave32 ---
__global__ __launch_bounds__(256) void ovo_main_kernel(
    const float* __restrict__ o1, const float* __restrict__ o2,
    const float* __restrict__ o3, const float* __restrict__ qin,
    const bf16*  __restrict__ Wt,
    float* __restrict__ ctx_out, float* __restrict__ attn_out) {
  extern __shared__ char smem[];
  bf16*  m_bf    = (bf16*)(smem + LDS_M);
  bf16*  q_bf    = (bf16*)(smem + LDS_Q);
  bf16*  tmp_bf  = (bf16*)(smem + LDS_TMP);
  bf16*  qT_bf   = (bf16*)(smem + LDS_QT);
  float* score   = (float*)(smem + LDS_SC);
  bf16*  attn_bf = (bf16*)(smem + LDS_ATT);

  const int b    = blockIdx.x;
  const int tid  = threadIdx.x;
  const size_t base = (size_t)b * (H_DIM * E_DIM);

  // ---------- Phase 0: load, fuse mean, convert to bf16, build qT ----------
  {
    const float4* p1 = (const float4*)(o1 + base);
    const float4* p2 = (const float4*)(o2 + base);
    const float4* p3 = (const float4*)(o3 + base);
    const float4* pq = (const float4*)(qin + base);
    const float third = 1.0f / 3.0f;
    #pragma unroll 4
    for (int i = tid; i < (H_DIM * E_DIM) / 4; i += 256) {
      float4 a = p1[i], c = p2[i], d = p3[i], qq = pq[i];
      int flat = i * 4;
      int row  = flat >> 9;           // h (0..63)
      int col  = flat & (E_DIM - 1);  // e (multiple of 4)
      bf16x4 mv, qv;
      mv.x = (bf16)((a.x + c.x + d.x) * third);
      mv.y = (bf16)((a.y + c.y + d.y) * third);
      mv.z = (bf16)((a.z + c.z + d.z) * third);
      mv.w = (bf16)((a.w + c.w + d.w) * third);
      qv.x = (bf16)qq.x; qv.y = (bf16)qq.y; qv.z = (bf16)qq.z; qv.w = (bf16)qq.w;
      *(bf16x4*)(m_bf + row * RS + col) = mv;
      *(bf16x4*)(q_bf + row * RS + col) = qv;
      qT_bf[(col + 0) * QT_RS + row] = qv.x;
      qT_bf[(col + 1) * QT_RS + row] = qv.y;
      qT_bf[(col + 2) * QT_RS + row] = qv.z;
      qT_bf[(col + 3) * QT_RS + row] = qv.w;
    }
  }
  __syncthreads();

  const int w    = tid >> 5;
  const int lane = tid & 31;
  const int lh   = lane >> 4;   // lane-half (A/B K-chunk selector)
  const int lr   = lane & 15;

  // ---------- Phase 1: tmp = m @ W  (64x512) ----------
  {
    const int mt = w >> 1;            // 4 M-tiles of 16 rows
    const int nh = w & 1;             // 2 N-halves of 16 tiles each
    const bf16* arow = m_bf + (mt * 16 + lr) * RS;
    for (int nt = nh * 16; nt < nh * 16 + 16; ++nt) {
      v8f acc = {};
      const bf16* bcol = Wt + (size_t)(nt * 16 + lr) * E_DIM;
      for (int k0 = 0; k0 < E_DIM; k0 += 32) {
        const bf16* ap = arow + k0 + lh * 8;
        v16bf A = load_frag(ap, ap + 16);
        const bf16* bp = bcol + k0 + lh * 8;
        v16bf Bv = load_frag(bp, bp + 16);
        acc = __builtin_amdgcn_wmma_f32_16x16x32_bf16(false, A, false, Bv,
                                                      (short)0, acc, false, false);
      }
      bf16* outc = tmp_bf + (mt * 16 + lh * 8) * RS + nt * 16 + lr;
      #pragma unroll
      for (int r = 0; r < 8; ++r) outc[r * RS] = (bf16)acc[r];
    }
  }
  __syncthreads();

  // ---------- Phase 2: score = tmp @ q^T  (64x64) ----------
  {
    for (int t = w * 2; t < w * 2 + 2; ++t) {
      const int mt2 = t >> 2, nt2 = t & 3;
      const bf16* arow = tmp_bf + (mt2 * 16 + lr) * RS;
      const bf16* brow = q_bf   + (nt2 * 16 + lr) * RS;  // B[k=f][n=g] = q[g][f]
      v8f acc = {};
      for (int k0 = 0; k0 < E_DIM; k0 += 32) {
        const bf16* ap = arow + k0 + lh * 8;
        v16bf A = load_frag(ap, ap + 16);
        const bf16* bp = brow + k0 + lh * 8;
        v16bf Bv = load_frag(bp, bp + 16);
        acc = __builtin_amdgcn_wmma_f32_16x16x32_bf16(false, A, false, Bv,
                                                      (short)0, acc, false, false);
      }
      float* sc = score + (mt2 * 16 + lh * 8) * SC_RS + nt2 * 16 + lr;
      #pragma unroll
      for (int r = 0; r < 8; ++r) sc[r * SC_RS] = acc[r];
    }
  }
  __syncthreads();

  // ---------- Phase 3: row softmax; write f32 attn out + bf16 copy ----------
  if (tid < H_DIM) {
    const int h = tid;
    float* row = score + h * SC_RS;
    float mx = row[0];
    #pragma unroll 8
    for (int g = 1; g < H_DIM; ++g) mx = fmaxf(mx, row[g]);
    float s = 0.f;
    #pragma unroll 8
    for (int g = 0; g < H_DIM; ++g) { float v = __expf(row[g] - mx); row[g] = v; s += v; }
    float inv = 1.0f / s;
    float* aout = attn_out + (size_t)b * (H_DIM * H_DIM) + (size_t)h * H_DIM;
    #pragma unroll 8
    for (int g = 0; g < H_DIM; ++g) {
      float v = row[g] * inv;
      aout[g] = v;
      attn_bf[h * AT_RS + g] = (bf16)v;
    }
  }
  __syncthreads();

  // ---------- Phase 4: context = attn @ q  (64x512) ----------
  {
    const int mt = w >> 1;
    const int nh = w & 1;
    const bf16* arow = attn_bf + (mt * 16 + lr) * AT_RS;
    float* crow = ctx_out + base + (size_t)(mt * 16 + lh * 8) * E_DIM;
    for (int nt = nh * 16; nt < nh * 16 + 16; ++nt) {
      v8f acc = {};
      const bf16* brow = qT_bf + (nt * 16 + lr) * QT_RS;  // B[k=g][n=e] = q[g][e]
      #pragma unroll
      for (int k0 = 0; k0 < H_DIM; k0 += 32) {
        const bf16* ap = arow + k0 + lh * 8;
        v16bf A = load_frag(ap, ap + 16);
        const bf16* bp = brow + k0 + lh * 8;
        v16bf Bv = load_frag(bp, bp + 16);
        acc = __builtin_amdgcn_wmma_f32_16x16x32_bf16(false, A, false, Bv,
                                                      (short)0, acc, false, false);
      }
      float* cp = crow + nt * 16 + lr;
      #pragma unroll
      for (int r = 0; r < 8; ++r) cp[r * E_DIM] = acc[r];
    }
  }
}

extern "C" void kernel_launch(void* const* d_in, const int* in_sizes, int n_in,
                              void* d_out, int out_size, void* d_ws, size_t ws_size,
                              hipStream_t stream) {
  const float* o1 = (const float*)d_in[0];
  const float* o2 = (const float*)d_in[1];
  const float* o3 = (const float*)d_in[2];
  const float* qm = (const float*)d_in[3];
  const float* W  = (const float*)d_in[4];

  bf16*  Wt   = (bf16*)d_ws;                       // 512*512*2 = 512 KB scratch
  float* ctx  = (float*)d_out;                     // (B,H,E) f32
  float* attn = ctx + (size_t)B_DIM * H_DIM * E_DIM;  // (B,H,H) f32

  wt_conv_kernel<<<(E_DIM * E_DIM) / 256, 256, 0, stream>>>(W, Wt);
  ovo_main_kernel<<<B_DIM, 256, LDS_TOTAL, stream>>>(o1, o2, o3, qm, Wt, ctx, attn);
}
